// RadianceRenderer_33921651703973
// MI455X (gfx1250) — matrix operand order repaired and checked by reading, not compiled, not run on hardware
//
#include <hip/hip_runtime.h>
#include <hip/hip_bf16.h>

typedef __attribute__((ext_vector_type(16))) _Float16 v16h;
typedef __attribute__((ext_vector_type(8)))  _Float16 v8h;
typedef __attribute__((ext_vector_type(8)))  float    v8f;

#define N_RAYS 8192
#define N_SAMP 64
#define HID 128
#define RAYS_PER_BLOCK 8
#define MLP_BLOCKS (N_RAYS / RAYS_PER_BLOCK)   // 1024
#define RAYDATA_F 20                            // tnear,tfar,dnorm,active,ynm[16]

// ---------------------------------------------------------------------------
// Kernel 0: per-ray setup — slab test + ||d|| + degree-3 spherical harmonics
// ---------------------------------------------------------------------------
__global__ __launch_bounds__(128) void ray_setup_kernel(
    const float* __restrict__ rays_o, const float* __restrict__ rays_d,
    const float* __restrict__ aabb, float* __restrict__ raydata)
{
    int r = blockIdx.x * blockDim.x + threadIdx.x;
    if (r >= N_RAYS) return;
    float ox = rays_o[3*r+0], oy = rays_o[3*r+1], oz = rays_o[3*r+2];
    float dx = rays_d[3*r+0], dy = rays_d[3*r+1], dz = rays_d[3*r+2];
    float lo0 = aabb[0], lo1 = aabb[1], lo2 = aabb[2];
    float hi0 = aabb[3], hi1 = aabb[4], hi2 = aabb[5];

    float tnear = 0.0f, tfar = 10.0f;
    {
        float inv = 1.0f/dx; float t1=(lo0-ox)*inv, t2=(hi0-ox)*inv;
        tnear = fmaxf(tnear, fminf(t1,t2)); tfar = fminf(tfar, fmaxf(t1,t2));
    }
    {
        float inv = 1.0f/dy; float t1=(lo1-oy)*inv, t2=(hi1-oy)*inv;
        tnear = fmaxf(tnear, fminf(t1,t2)); tfar = fminf(tfar, fmaxf(t1,t2));
    }
    {
        float inv = 1.0f/dz; float t1=(lo2-oz)*inv, t2=(hi2-oz)*inv;
        tnear = fmaxf(tnear, fminf(t1,t2)); tfar = fminf(tfar, fmaxf(t1,t2));
    }
    bool active = tnear < tfar;
    if (!active) { tnear = 0.0f; tfar = 1.0f; }

    float dn = sqrtf(dx*dx + dy*dy + dz*dz);
    float x = dx/dn, y = dy/dn, z = dz/dn;
    float x2 = x*x, y2 = y*y, z2 = z*z;

    float* out = raydata + r*RAYDATA_F;
    out[0] = tnear; out[1] = tfar; out[2] = dn; out[3] = active ? 1.0f : 0.0f;
    out[4]  = 0.282094791773878f;
    out[5]  = -0.488602511902920f * y;
    out[6]  =  0.488602511902920f * z;
    out[7]  = -0.488602511902920f * x;
    out[8]  =  1.092548430592079f * x * y;
    out[9]  = -1.092548430592079f * y * z;
    out[10] =  0.315391565252520f * (3.0f*z2 - 1.0f);
    out[11] = -1.092548430592079f * x * z;
    out[12] =  0.546274215296040f * (x2 - y2);
    out[13] = -0.590043589926644f * y * (3.0f*x2 - y2);
    out[14] =  2.890611442640554f * x * y * z;
    out[15] = -0.457045799464466f * y * (5.0f*z2 - 1.0f);
    out[16] =  0.373176332590115f * z * (5.0f*z2 - 3.0f);
    out[17] = -0.457045799464466f * x * (5.0f*z2 - 1.0f);
    out[18] =  1.445305721320277f * z * (x2 - y2);
    out[19] = -0.590043589926644f * x * (x2 - 3.0f*y2);
}

// ---------------------------------------------------------------------------
// Kernel 1: fused radiance-field MLP, WMMA f16->f32.
// 128 threads = 4 waves; wave w handles samples [16w,16w+16) of one ray.
// ---------------------------------------------------------------------------
__global__ __launch_bounds__(128) void mlp_kernel(
    const float* __restrict__ rays_o, const float* __restrict__ rays_d,
    const float* __restrict__ aabb,
    const float* __restrict__ W1, const float* __restrict__ b1,
    const float* __restrict__ W2, const float* __restrict__ b2,
    const float* __restrict__ Wd, const float* __restrict__ bd,
    const float* __restrict__ Wf, const float* __restrict__ bf,
    const float* __restrict__ Wc1, const float* __restrict__ bc1,
    const float* __restrict__ Wc2, const float* __restrict__ bc2,
    const float* __restrict__ raydata, float* __restrict__ ptdata)
{
    // LDS: ~61.8 KB total
    __shared__ __attribute__((aligned(32))) _Float16 sW2B[32*32*16]; // 32 B-frags, swizzled
    __shared__ __attribute__((aligned(32))) _Float16 sWc1B[8*32*16]; // 8 B-frags, swizzled
    __shared__ __attribute__((aligned(32))) _Float16 sActAll[4*16*HID];
    __shared__ __attribute__((aligned(32))) _Float16 sWc2T[3*HID];   // transposed [c][k]
    __shared__ float sW1[3*HID];
    __shared__ float sB1[HID], sB2[HID], sWd[HID], sBc1[HID];
    __shared__ float sBf[16];
    __shared__ float sMisc[12]; // lo0..2, sc0..2, bd, bc2[0..2]

    const int tid = threadIdx.x;
    const int ln  = tid & 31;       // lane
    const int w   = tid >> 5;       // wave in block
    const int m   = ln & 15;        // row (A/C) or col (B/C) within 16x16 tile
    const int aoff = (ln >= 16) ? 8 : 0;   // A-layout / C-row offset for high lanes

    // ---- cooperative weight preload (pre-swizzled to WMMA B-fragment order)
    for (int i = tid; i < 32*32*16; i += 128) {
        int f = i >> 9, l2 = (i >> 4) & 31, j = i & 15;
        int nt = f >> 2, kb = f & 3;
        int n = nt*16 + (l2 & 15);
        int k = kb*32 + ((l2 >= 16) ? 8 : 0) + ((j < 8) ? j : (j + 8));
        sW2B[i] = (_Float16)W2[k*HID + n];
    }
    for (int i = tid; i < 8*32*16; i += 128) {
        int f = i >> 9, l2 = (i >> 4) & 31, j = i & 15;
        int n = f*16 + (l2 & 15);
        int k = ((l2 >= 16) ? 8 : 0) + ((j < 8) ? j : (j + 8));
        sWc1B[i] = (_Float16)Wc1[k*HID + n];
    }
    for (int i = tid; i < 3*HID; i += 128) sW1[i] = W1[i];
    if (tid < HID) { sB1[tid]=b1[tid]; sB2[tid]=b2[tid]; sWd[tid]=Wd[tid]; sBc1[tid]=bc1[tid]; }
    if (tid < 16)  sBf[tid] = bf[tid];
    for (int i = tid; i < 3*HID; i += 128) {
        int c = i >> 7, k = i & 127;
        sWc2T[i] = (_Float16)Wc2[k*3 + c];
    }
    if (tid == 0) {
        sMisc[0]=aabb[0]; sMisc[1]=aabb[1]; sMisc[2]=aabb[2];
        sMisc[3]=2.0f/(aabb[3]-aabb[0]);
        sMisc[4]=2.0f/(aabb[4]-aabb[1]);
        sMisc[5]=2.0f/(aabb[5]-aabb[2]);
        sMisc[6]=bd[0]; sMisc[7]=bc2[0]; sMisc[8]=bc2[1]; sMisc[9]=bc2[2];
    }

    // Wf B-fragments live in registers for the whole kernel (N=16 -> 1 tile)
    v16h wfB[4];
#pragma unroll
    for (int kb = 0; kb < 4; ++kb)
#pragma unroll
        for (int j = 0; j < 16; ++j) {
            int k = kb*32 + ((ln >= 16) ? 8 : 0) + ((j < 8) ? j : (j + 8));
            wfB[kb][j] = (_Float16)Wf[k*16 + m];
        }

    __syncthreads();

    const float lo0=sMisc[0], lo1=sMisc[1], lo2=sMisc[2];
    const float sc0=sMisc[3], sc1=sMisc[4], sc2=sMisc[5];
    const float bdv=sMisc[6];
    _Float16* act = &sActAll[w * 16 * HID];

    for (int it = 0; it < RAYS_PER_BLOCK; ++it) {
        const int ray = blockIdx.x * RAYS_PER_BLOCK + it;
        const float* rd = raydata + ray * RAYDATA_F;
        const float tn = rd[0], tf = rd[1];
        const float step = (tf - tn) * (1.0f / (float)N_SAMP);
        const int t = w*16 + m;
        const float ts = tn + ((float)t + 0.5f) * step;

        const float ox = rays_o[3*ray+0], oy = rays_o[3*ray+1], oz = rays_o[3*ray+2];
        const float dx = rays_d[3*ray+0], dy = rays_d[3*ray+1], dz = rays_d[3*ray+2];
        const float nx = (ox + dx*ts - lo0)*sc0 - 1.0f;
        const float ny = (oy + dy*ts - lo1)*sc1 - 1.0f;
        const float nz = (oz + dz*ts - lo2)*sc2 - 1.0f;
        const float inside =
            (fabsf(nx) <= 1.0f && fabsf(ny) <= 1.0f && fabsf(nz) <= 1.0f) ? 1.0f : 0.0f;

        // ---- layer 1 (3->128), computed straight into A-fragment registers
        v16h aF[4];
#pragma unroll
        for (int kb = 0; kb < 4; ++kb)
#pragma unroll
            for (int j = 0; j < 16; ++j) {
                int k = kb*32 + aoff + ((j < 8) ? j : (j + 8));
                float v = nx*sW1[k] + ny*sW1[HID+k] + nz*sW1[2*HID+k] + sB1[k];
                aF[kb][j] = (_Float16)fmaxf(v, 0.0f);
            }

        // ---- layer 2 (128->128) : 8 N-tiles x 4 K-steps of WMMA
#pragma unroll
        for (int nt = 0; nt < 8; ++nt) {
            v8f acc = {};
#pragma unroll
            for (int kb = 0; kb < 4; ++kb) {
                v16h bB = *(const v16h*)&sW2B[((nt*4 + kb)*32 + ln) * 16];
                acc = __builtin_amdgcn_wmma_f32_16x16x32_f16(
                        false, aF[kb], false, bB, (short)0, acc, false, false);
            }
            float bias = sB2[nt*16 + m];
#pragma unroll
            for (int v = 0; v < 8; ++v) {
                float hv = fmaxf(acc[v] + bias, 0.0f);
                act[(v + aoff)*HID + nt*16 + m] = (_Float16)hv;
            }
        }
        asm volatile("s_wait_dscnt 0x0" ::: "memory");

        // ---- reload h2 as A-fragments
        v16h a2[4];
#pragma unroll
        for (int kb = 0; kb < 4; ++kb)
#pragma unroll
            for (int j = 0; j < 16; ++j) {
                int k = kb*32 + aoff + ((j < 8) ? j : (j + 8));
                a2[kb][j] = act[m*HID + k];
            }

        // ---- sigma = exp(h2 . Wd + bd)  (lanes 0..15, one row each)
        float sg = 0.0f;
        if (ln < 16) {
            float s = bdv;
#pragma unroll 4
            for (int kk = 0; kk < 16; ++kk) {
                v8h hv = *(const v8h*)&act[ln*HID + kk*8];
#pragma unroll
                for (int q = 0; q < 8; ++q) s += (float)hv[q] * sWd[kk*8 + q];
            }
            sg = __expf(s);
        }

        // ---- feat = h2 @ Wf + bf (linear), written into cat cols [0,16)
        v8f fa = {};
#pragma unroll
        for (int kb = 0; kb < 4; ++kb)
            fa = __builtin_amdgcn_wmma_f32_16x16x32_f16(
                    false, a2[kb], false, wfB[kb], (short)0, fa, false, false);
        {
            float fb = sBf[m];
            asm volatile("" ::: "memory");
#pragma unroll
            for (int v = 0; v < 8; ++v)
                act[(v + aoff)*HID + m] = (_Float16)(fa[v] + fb);
        }
        // ---- ynm into cat cols [16,32)
        if (ln < 16) {
#pragma unroll
            for (int j = 0; j < 16; ++j)
                act[ln*HID + 16 + j] = (_Float16)rd[4 + j];
        }
        asm volatile("s_wait_dscnt 0x0" ::: "memory");

        // ---- cat A-fragment (K=32 exactly)
        v16h ca;
#pragma unroll
        for (int j = 0; j < 16; ++j) {
            int k = aoff + ((j < 8) ? j : (j + 8));
            ca[j] = act[m*HID + k];
        }

        // ---- hc = relu(cat @ Wc1 + bc1) : 8 WMMAs (one K-step each)
#pragma unroll
        for (int nt = 0; nt < 8; ++nt) {
            v8f hacc = {};
            v16h bB = *(const v16h*)&sWc1B[(nt*32 + ln) * 16];
            hacc = __builtin_amdgcn_wmma_f32_16x16x32_f16(
                    false, ca, false, bB, (short)0, hacc, false, false);
            float bias = sBc1[nt*16 + m];
#pragma unroll
            for (int v = 0; v < 8; ++v)
                act[(v + aoff)*HID + nt*16 + m] = (_Float16)fmaxf(hacc[v] + bias, 0.0f);
        }
        asm volatile("s_wait_dscnt 0x0" ::: "memory");

        // ---- color = sigmoid(hc @ Wc2 + bc2); emit per-point record
        if (ln < 16) {
            float s0 = sMisc[7], s1 = sMisc[8], s2 = sMisc[9];
#pragma unroll 2
            for (int kk = 0; kk < 16; ++kk) {
                v8h hv = *(const v8h*)&act[ln*HID + kk*8];
                v8h w0 = *(const v8h*)&sWc2T[0*HID + kk*8];
                v8h w1 = *(const v8h*)&sWc2T[1*HID + kk*8];
                v8h w2 = *(const v8h*)&sWc2T[2*HID + kk*8];
#pragma unroll
                for (int q = 0; q < 8; ++q) {
                    float h = (float)hv[q];
                    s0 += h * (float)w0[q];
                    s1 += h * (float)w1[q];
                    s2 += h * (float)w2[q];
                }
            }
            float cr = 1.0f / (1.0f + __expf(-s0));
            float cg = 1.0f / (1.0f + __expf(-s1));
            float cb = 1.0f / (1.0f + __expf(-s2));
            float4 rec;
            rec.x = sg * inside;
            rec.y = cr * inside;
            rec.z = cg * inside;
            rec.w = cb * inside;
            *(float4*)&ptdata[(size_t)(ray*N_SAMP + t) * 4] = rec;
        }
    }
}

// ---------------------------------------------------------------------------
// Kernel 2: per-ray transmittance integration (exclusive cumsum of tau)
// ---------------------------------------------------------------------------
__global__ __launch_bounds__(128) void integrate_kernel(
    const float* __restrict__ raydata, const float* __restrict__ ptdata,
    float* __restrict__ out)
{
    int r = blockIdx.x * blockDim.x + threadIdx.x;
    if (r >= N_RAYS) return;
    const float* rd = raydata + r*RAYDATA_F;
    const float tn = rd[0], tf = rd[1], dn = rd[2], am = rd[3];
    const float step = (tf - tn) * (1.0f / (float)N_SAMP);

    float excl = 0.0f, cr = 0.0f, cg = 0.0f, cb = 0.0f, wa = 0.0f, wd = 0.0f;
    for (int t = 0; t < N_SAMP; ++t) {
        float4 p = *(const float4*)&ptdata[(size_t)(r*N_SAMP + t) * 4];
        float delta = (t < N_SAMP-1) ? step : (0.5f*step + 1.0f); // BOOSTER=1
        float ts = tn + ((float)t + 0.5f) * step;
        float tau = p.x * delta * dn;
        float wgt = __expf(-excl) * (1.0f - __expf(-tau));
        cr += wgt * p.y; cg += wgt * p.z; cb += wgt * p.w;
        wa += wgt; wd += wgt * ts;
        excl += tau;
    }
    out[r*5+0] = cr * am;
    out[r*5+1] = cg * am;
    out[r*5+2] = cb * am;
    out[r*5+3] = wa * am;
    out[r*5+4] = wd * am;
}

// ---------------------------------------------------------------------------
extern "C" void kernel_launch(void* const* d_in, const int* in_sizes, int n_in,
                              void* d_out, int out_size, void* d_ws, size_t ws_size,
                              hipStream_t stream) {
    const float* rays_o = (const float*)d_in[0];
    const float* rays_d = (const float*)d_in[1];
    const float* aabb   = (const float*)d_in[2];
    const float* W1  = (const float*)d_in[3];
    const float* b1  = (const float*)d_in[4];
    const float* W2  = (const float*)d_in[5];
    const float* b2  = (const float*)d_in[6];
    const float* Wd  = (const float*)d_in[7];
    const float* bd  = (const float*)d_in[8];
    const float* Wf  = (const float*)d_in[9];
    const float* bf  = (const float*)d_in[10];
    const float* Wc1 = (const float*)d_in[11];
    const float* bc1 = (const float*)d_in[12];
    const float* Wc2 = (const float*)d_in[13];
    const float* bc2 = (const float*)d_in[14];
    float* out = (float*)d_out;

    float* raydata = (float*)d_ws;                                    // 8192*20 f32
    float* ptdata  = (float*)((char*)d_ws + (size_t)N_RAYS*RAYDATA_F*sizeof(float));

    ray_setup_kernel<<<N_RAYS/128, 128, 0, stream>>>(rays_o, rays_d, aabb, raydata);
    mlp_kernel<<<MLP_BLOCKS, 128, 0, stream>>>(rays_o, rays_d, aabb,
                                               W1, b1, W2, b2, Wd, bd, Wf, bf,
                                               Wc1, bc1, Wc2, bc2, raydata, ptdata);
    integrate_kernel<<<N_RAYS/128, 128, 0, stream>>>(raydata, ptdata, out);
}
